// GraphAttentionLayer_75849122448209
// MI455X (gfx1250) — matrix operand order repaired and checked by reading, not compiled, not run on hardware
//
#include <hip/hip_runtime.h>

// Problem constants (fixed by the reference setup)
#define BB 2
#define NN 2048
#define DD 512      // D = H*KH
#define HH 8
#define KH 64
#define NEG_INF_S  (-1.0e30f)
#define NEG_SLOPE_S (0.2f)

typedef __attribute__((ext_vector_type(16))) __bf16          v16bf;
typedef __attribute__((ext_vector_type(16))) unsigned short  v16u;
typedef __attribute__((ext_vector_type(8)))  float           v8f;

// Native bf16 conversion: let the compiler pick the packed cvt instructions.
__device__ __forceinline__ __bf16 f2bf(float f) { return (__bf16)f; }
__device__ __forceinline__ float bf2f(unsigned short h) {
  return __uint_as_float(((unsigned int)h) << 16);
}

// ---------------------------------------------------------------------------
// Kernel 0: W_proj [Din,Dout] f32 -> WT [Dout,Din] bf16 (transposed for the
// WMMA B-operand: lane = output column, K contiguous per lane).
// ---------------------------------------------------------------------------
__global__ void wt_prep(const float* __restrict__ W, __bf16* __restrict__ WT) {
  int idx = blockIdx.x * blockDim.x + threadIdx.x;   // 512*512 threads
  int i = idx >> 9;        // Din row
  int o = idx & 511;       // Dout col
  WT[(size_t)o * DD + i] = f2bf(W[(size_t)i * DD + o]);
}

// ---------------------------------------------------------------------------
// Kernel 1: xp = x @ W_proj via bf16 WMMA (f32 accum).
// One wave per 16x16 tile of the [4096 x 512] output.
// Writes xh[b,h,n,kd] and xhT[b,h,kd,n] in bf16.
// ---------------------------------------------------------------------------
__global__ void proj_gemm(const float* __restrict__ x,
                          const __bf16* __restrict__ WT,
                          __bf16* __restrict__ xh,
                          __bf16* __restrict__ xhT) {
  const int wave = (blockIdx.x * blockDim.x + threadIdx.x) >> 5;
  const int lane = threadIdx.x & 31;
  const int g  = lane >> 4;      // lane group (K-half selector)
  const int lr = lane & 15;      // A: row M ; B: column N
  const int to = wave & 31;      // 512/16 = 32 column tiles
  const int tm = wave >> 5;      // 4096/16 = 256 row tiles

  const float*  xrow = x  + (size_t)(tm * 16 + lr) * DD;
  const __bf16* wcol = WT + (size_t)(to * 16 + lr) * DD;

  v8f acc = {};
  #pragma unroll 4
  for (int k0 = 0; k0 < DD; k0 += 32) {
    // A operand: K = k0 + 16g + e contiguous
    float4 f0 = *(const float4*)(xrow + k0 + 16 * g);
    float4 f1 = *(const float4*)(xrow + k0 + 16 * g + 4);
    float4 f2 = *(const float4*)(xrow + k0 + 16 * g + 8);
    float4 f3 = *(const float4*)(xrow + k0 + 16 * g + 12);
    v16bf a;
    a[0]  = f2bf(f0.x); a[1]  = f2bf(f0.y); a[2]  = f2bf(f0.z); a[3]  = f2bf(f0.w);
    a[4]  = f2bf(f1.x); a[5]  = f2bf(f1.y); a[6]  = f2bf(f1.z); a[7]  = f2bf(f1.w);
    a[8]  = f2bf(f2.x); a[9]  = f2bf(f2.y); a[10] = f2bf(f2.z); a[11] = f2bf(f2.w);
    a[12] = f2bf(f3.x); a[13] = f2bf(f3.y); a[14] = f2bf(f3.z); a[15] = f2bf(f3.w);
    // B operand: lane holds column lr, K = k0 + 16g + (0..15) contiguous
    v16bf b = *(const v16bf*)(wcol + k0 + 16 * g);
    acc = __builtin_amdgcn_wmma_f32_16x16x32_bf16(false, a, false, b,
                                                  (short)0, acc, false, false);
  }

  // C layout: VGPR r, lane -> row = r + 8*(lane/16), col = lane&15
  const int ocol = to * 16 + lr;
  const int h  = ocol >> 6;
  const int kd = ocol & 63;
  #pragma unroll
  for (int r = 0; r < 8; ++r) {
    const int nrow = tm * 16 + r + 8 * g;
    const int b = nrow >> 11;
    const int n = nrow & 2047;
    __bf16 bv = f2bf(acc[r]);
    xh [(((size_t)(b * HH + h)) * NN + n) * KH + kd] = bv;
    xhT[(((size_t)(b * HH + h)) * KH + kd) * NN + n] = bv;
  }
}

// ---------------------------------------------------------------------------
// Kernel 2: a_src/a_dst per (b,h,n) — tiny dots over k=64 (vectorized loads).
// ---------------------------------------------------------------------------
__global__ void att_coef(const unsigned short* __restrict__ xh,
                         const float* __restrict__ W_att,
                         float* __restrict__ a_src, float* __restrict__ a_dst) {
  int idx = blockIdx.x * blockDim.x + threadIdx.x;  // B*H*N = 32768
  int h = (idx >> 11) & 7;
  const unsigned short* row = xh + (size_t)idx * KH;
  const float* w = W_att + h * 2 * KH;
  float s = 0.0f, d = 0.0f;
  #pragma unroll
  for (int t0 = 0; t0 < KH; t0 += 16) {
    v16u v = *(const v16u*)(row + t0);
    #pragma unroll
    for (int t = 0; t < 16; ++t) {
      float xv = bf2f(v[t]);
      s += xv * w[t0 + t];
      d += xv * w[KH + t0 + t];
    }
  }
  a_src[idx] = s;
  a_dst[idx] = d;
}

// ---------------------------------------------------------------------------
// Kernel 3: fused masked-softmax attention + AV GEMM (flash style).
// One wave per (b, h, 16-row tile). P kept in registers in the WMMA A layout,
// 4x v_wmma_f32_16x16x32_bf16 per 32-column chunk.
// ---------------------------------------------------------------------------
__global__ void gat_flash(const int* __restrict__ A,
                          const __bf16* __restrict__ xhT,
                          const float* __restrict__ a_src,
                          const float* __restrict__ a_dst,
                          const float* __restrict__ b_att,
                          float* __restrict__ out) {
  const int wave = (blockIdx.x * blockDim.x + threadIdx.x) >> 5;
  const int lane = threadIdx.x & 31;
  const int g  = lane >> 4;
  const int lr = lane & 15;
  const int itile = wave & ((NN / 16) - 1);   // 128 row tiles
  const int bh    = wave >> 7;
  const int b = bh >> 3;
  const int h = bh & 7;

  const int irow = itile * 16 + lr;           // this lane's score row
  const float asrc_i = a_src[bh * NN + irow];
  const float bias   = b_att[h];
  const int*    Arow = A     + ((size_t)b * NN + irow) * NN;
  const float*  drow = a_dst + (size_t)bh * NN;
  const __bf16* xT   = xhT   + (size_t)bh * KH * NN;

  float m = -3.0e38f, l = 0.0f;
  v8f acc0 = {}, acc1 = {}, acc2 = {}, acc3 = {};

  for (int j0 = 0; j0 < NN; j0 += 32) {
    // Speculative prefetch of next chunk's mask row (global_prefetch_b8).
    if (j0 + 32 < NN) {
      __builtin_prefetch(Arow + j0 + 32, 0, 1);
      __builtin_prefetch(drow + j0 + 32, 0, 1);
    }
    // A-operand lane pattern: j = j0 + 16g + e contiguous
    int4 m0 = *(const int4*)(Arow + j0 + 16 * g);
    int4 m1 = *(const int4*)(Arow + j0 + 16 * g + 4);
    int4 m2 = *(const int4*)(Arow + j0 + 16 * g + 8);
    int4 m3 = *(const int4*)(Arow + j0 + 16 * g + 12);
    float4 d0 = *(const float4*)(drow + j0 + 16 * g);
    float4 d1 = *(const float4*)(drow + j0 + 16 * g + 4);
    float4 d2 = *(const float4*)(drow + j0 + 16 * g + 8);
    float4 d3 = *(const float4*)(drow + j0 + 16 * g + 12);

    int   mk[16] = {m0.x,m0.y,m0.z,m0.w, m1.x,m1.y,m1.z,m1.w,
                    m2.x,m2.y,m2.z,m2.w, m3.x,m3.y,m3.z,m3.w};
    float dv[16] = {d0.x,d0.y,d0.z,d0.w, d1.x,d1.y,d1.z,d1.w,
                    d2.x,d2.y,d2.z,d2.w, d3.x,d3.y,d3.z,d3.w};

    float s[16];
    float cmax = -3.0e38f;
    #pragma unroll
    for (int e = 0; e < 16; ++e) {
      float sc = asrc_i + dv[e] + bias;
      sc = (sc > 0.0f) ? sc : NEG_SLOPE_S * sc;      // leaky relu
      sc = (mk[e] > 0) ? sc : NEG_INF_S;             // adjacency mask
      s[e] = sc;
      cmax = fmaxf(cmax, sc);
    }
    // row = lr is shared by lanes l and l^16 (each holding half the j's)
    cmax = fmaxf(cmax, __shfl_xor(cmax, 16));
    float mn    = fmaxf(m, cmax);
    float scale = __expf(m - mn);

    float psum = 0.0f;
    v16bf a;
    #pragma unroll
    for (int e = 0; e < 16; ++e) {
      float p = __expf(s[e] - mn);
      psum += p;
      a[e] = f2bf(p);
    }
    psum += __shfl_xor(psum, 16);
    l = l * scale + psum;
    m = mn;

    // rescale accumulators: acc VGPR r holds row r+8g; lane (r+8g) owns that row's scale
    float rs[8];
    #pragma unroll
    for (int r = 0; r < 8; ++r) rs[r] = __shfl(scale, r + 8 * g);
    #pragma unroll
    for (int r = 0; r < 8; ++r) {
      acc0[r] *= rs[r]; acc1[r] *= rs[r]; acc2[r] *= rs[r]; acc3[r] *= rs[r];
    }

    // B tiles: lane = column kd (within 16-chunk), K = j0 + 16g + (0..15) contiguous in xhT
    v16bf b0 = *(const v16bf*)(xT + (size_t)( 0 + lr) * NN + j0 + 16 * g);
    v16bf b1 = *(const v16bf*)(xT + (size_t)(16 + lr) * NN + j0 + 16 * g);
    v16bf b2 = *(const v16bf*)(xT + (size_t)(32 + lr) * NN + j0 + 16 * g);
    v16bf b3 = *(const v16bf*)(xT + (size_t)(48 + lr) * NN + j0 + 16 * g);
    acc0 = __builtin_amdgcn_wmma_f32_16x16x32_bf16(false, a, false, b0, (short)0, acc0, false, false);
    acc1 = __builtin_amdgcn_wmma_f32_16x16x32_bf16(false, a, false, b1, (short)0, acc1, false, false);
    acc2 = __builtin_amdgcn_wmma_f32_16x16x32_bf16(false, a, false, b2, (short)0, acc2, false, false);
    acc3 = __builtin_amdgcn_wmma_f32_16x16x32_bf16(false, a, false, b3, (short)0, acc3, false, false);
  }

  // epilogue: divide by row sum, relu, scatter to [B, N, H*k]
  float li[8];
  #pragma unroll
  for (int r = 0; r < 8; ++r) li[r] = __shfl(l, r + 8 * g);
  #pragma unroll
  for (int r = 0; r < 8; ++r) {
    const float inv = 1.0f / li[r];
    const int row = itile * 16 + r + 8 * g;
    float* orow = out + ((size_t)b * NN + row) * DD + h * KH;
    orow[ 0 + lr] = fmaxf(acc0[r] * inv, 0.0f);
    orow[16 + lr] = fmaxf(acc1[r] * inv, 0.0f);
    orow[32 + lr] = fmaxf(acc2[r] * inv, 0.0f);
    orow[48 + lr] = fmaxf(acc3[r] * inv, 0.0f);
  }
}

// ---------------------------------------------------------------------------
extern "C" void kernel_launch(void* const* d_in, const int* in_sizes, int n_in,
                              void* d_out, int out_size, void* d_ws, size_t ws_size,
                              hipStream_t stream) {
  const float* x      = (const float*)d_in[0];
  const int*   A      = (const int*)  d_in[1];
  const float* W_proj = (const float*)d_in[2];
  const float* W_att  = (const float*)d_in[3];
  const float* b_att  = (const float*)d_in[4];
  float* out = (float*)d_out;

  // Workspace layout (all 32B-aligned offsets):
  //   WT   bf16 [512][512]            : 524288 B  @ 0
  //   xh   bf16 [B][H][N][64]         : 4 MiB     @ 524288
  //   xhT  bf16 [B][H][64][N]         : 4 MiB     @ 4718592
  //   a_src f32 [B*H*N]               : 128 KiB   @ 8912896
  //   a_dst f32 [B*H*N]               : 128 KiB   @ 9043968
  char* ws = (char*)d_ws;
  __bf16* WT   = (__bf16*)(ws);
  __bf16* xh   = (__bf16*)(ws + 524288);
  __bf16* xhT  = (__bf16*)(ws + 4718592);
  float*  asrc = (float*)(ws + 8912896);
  float*  adst = (float*)(ws + 9043968);

  wt_prep  <<<(DD * DD) / 256, 256, 0, stream>>>(W_proj, WT);
  proj_gemm<<<1024, 256, 0, stream>>>(x, WT, xh, xhT);           // 8192 waves = 256x32 tiles
  att_coef <<<(BB * HH * NN) / 256, 256, 0, stream>>>((const unsigned short*)xh, W_att, asrc, adst);
  gat_flash<<<512, 128, 0, stream>>>(A, xhT, asrc, adst, b_att, out); // 2048 waves
}
